// BiLSTM_55525337203270
// MI455X (gfx1250) — matrix operand order repaired and checked by reading, not compiled, not run on hardware
//
#include <hip/hip_runtime.h>
#include <hip/hip_bf16.h>
#include <cstdint>
#include <cstddef>

typedef __attribute__((ext_vector_type(16))) __bf16 v16bf;
typedef __attribute__((ext_vector_type(8)))  __bf16 v8bf;
typedef __attribute__((ext_vector_type(8)))  float  v8f;
typedef unsigned int u32x4 __attribute__((ext_vector_type(4)));
typedef int          i32x4 __attribute__((ext_vector_type(4)));
typedef int          i32x8 __attribute__((ext_vector_type(8)));

#define TT   500
#define BB   256
#define INP  203
#define KX   224   // padded x width (7*32)
#define HID  128
#define G4   512   // 4*HID
#define EMBW 200
#define NTM  16    // M(batch) tiles per timestep
#define NTN  32    // N tiles (512/16)

static __device__ __forceinline__ float sigf(float x) {
  return 1.0f / (1.0f + __expf(-x));
}
static __device__ __forceinline__ float tanh_fast(float x) {
  float e = __expf(2.0f * x);
  return 1.0f - 2.0f / (e + 1.0f);   // robust at +/-inf
}
static __device__ __forceinline__ v16bf compose16(v8bf lo, v8bf hi) {
  v16bf r;
#pragma unroll
  for (int i = 0; i < 8; ++i) { r[i] = lo[i]; r[i + 8] = hi[i]; }
  return r;
}

// ---------------- embedding gather -> X bf16 [T*B][224] ----------------
__global__ void embed_kernel(const int* __restrict__ words, const int* __restrict__ caps,
                             const float* __restrict__ wemb, const float* __restrict__ cemb,
                             __bf16* __restrict__ X) {
  int m = blockIdx.x;            // m = t*BB + b
  int b = m & (BB - 1);
  int t = m >> 8;
  int k = threadIdx.x;           // 0..223
  float v = 0.0f;
  if (k < EMBW) {
    int w = words[(size_t)b * TT + t];
    v = wemb[(size_t)w * EMBW + k];
  } else if (k < INP) {
    int c = caps[(size_t)b * TT + t];
    v = cemb[c * 3 + (k - EMBW)];
  }
  X[(size_t)m * KX + k] = (__bf16)v;
}

// ------- transpose+pad weights: W[331][512] f32 -> Wtx[512][224], Wth[512][128] bf16 -------
__global__ void prepw_kernel(const float* __restrict__ W, __bf16* __restrict__ Wtx,
                             __bf16* __restrict__ Wth) {
  int idx = blockIdx.x * blockDim.x + threadIdx.x;
  if (idx >= G4 * (KX + HID)) return;
  int n = idx / (KX + HID);
  int k = idx % (KX + HID);
  if (k < KX) {
    float v = (k < INP) ? W[(size_t)k * G4 + n] : 0.0f;
    Wtx[(size_t)n * KX + k] = (__bf16)v;
  } else {
    int kh = k - KX;
    Wth[(size_t)n * HID + kh] = (__bf16)W[(size_t)(INP + kh) * G4 + n];
  }
}

// ------- Gx = X @ Wx + b : one wave computes a 16(M) x 128(N) strip, K=224 -------
// Output stored in WMMA C-fragment layout: [t][nt][mt][lane][8] floats.
__global__ void __launch_bounds__(256) gemmx_kernel(const __bf16* __restrict__ X,
                             const __bf16* __restrict__ Wtx,
                             const float* __restrict__ bias,
                             float* __restrict__ Gx) {
  int lane  = threadIdx.x & 31;
  int gw    = blockIdx.x * 8 + (threadIdx.x >> 5);
  int mtile = gw >> 2;           // 0..7999
  int ngrp  = gw & 3;            // which 128-wide N group
  int hi    = (lane >= 16);
  int row   = mtile * 16 + (lane & 15);
  const __bf16* xrow = X + (size_t)row * KX;

  v8f acc[8] = {};
#pragma unroll
  for (int kc = 0; kc < 7; ++kc) {
    int c0 = kc * 32 + (hi ? 16 : 0);
    v8bf a0 = *(const v8bf*)(xrow + c0);
    v8bf a1 = *(const v8bf*)(xrow + c0 + 8);
    v16bf a = compose16(a0, a1);
    int ks = kc * 32 + (hi ? 16 : 0);
#pragma unroll
    for (int nt2 = 0; nt2 < 8; ++nt2) {
      int n = ngrp * 128 + nt2 * 16 + (lane & 15);
      v16bf bf = *(const v16bf*)(Wtx + (size_t)n * KX + ks);
      acc[nt2] = __builtin_amdgcn_wmma_f32_16x16x32_bf16(
          false, a, false, bf, (short)0, acc[nt2], false, false);
    }
  }
  int t  = mtile >> 4;
  int mt = mtile & 15;
#pragma unroll
  for (int nt2 = 0; nt2 < 8; ++nt2) {
    int nt = ngrp * 8 + nt2;
    float bv = bias[nt * 16 + (lane & 15)];
#pragma unroll
    for (int r = 0; r < 8; ++r) acc[nt2][r] += bv;
    *(v8f*)(Gx + (((size_t)t * NTN + nt) * NTM + mt) * 256 + lane * 8) = acc[nt2];
  }
}

// ---- TDM: prefetch one timestep's 32 gate tiles (32 x 1KB rows, 16KB apart) into LDS ----
static __device__ __forceinline__ void tdm_issue(const float* Gx, int t, int bt,
                                                 uint32_t ldsoff) {
  uint64_t ga = (uint64_t)(uintptr_t)Gx + ((uint64_t)t * NTN * NTM + bt) * 1024u;
  u32x4 g0;
  g0[0] = 1u;                                               // count=1 (valid user D#)
  g0[1] = ldsoff;                                           // lds_addr (bytes)
  g0[2] = (uint32_t)ga;                                     // global_addr[31:0]
  g0[3] = (uint32_t)((ga >> 32) & 0x01FFFFFFu) | (2u << 30);// global_addr[56:32] | type=2
  i32x8 g1;
  g1[0] = 0x00020000;          // data_size=4B, workgroup_mask=0, no iterate/pad/barrier
  g1[1] = (int)(256u << 16);   // tensor_dim0 = 256 elements (bits 63:48 = low16)
  g1[2] = (int)(32u << 16);    // tensor_dim0 hi16 = 0 | tensor_dim1 low16 = 32
  g1[3] = (int)(256u << 16);   // tensor_dim1 hi16 = 0 | tile_dim0 = 256
  g1[4] = 32;                  // tile_dim1 = 32, tile_dim2 = 0
  g1[5] = 4096;                // tensor_dim0_stride = 4096 elements (16KB)
  g1[6] = 0;                   // stride hi | tensor_dim1_stride lo
  g1[7] = 0;
  i32x4 z4 = {0, 0, 0, 0};
#if defined(__clang_major__) && __clang_major__ >= 23
  i32x8 z8 = {0, 0, 0, 0, 0, 0, 0, 0};
  __builtin_amdgcn_tensor_load_to_lds(g0, g1, z4, z4, z8, 0);
#else
  __builtin_amdgcn_tensor_load_to_lds(g0, g1, z4, z4, 0);
#endif
}

// ------- persistent recurrence: batch rows are independent -> 16 blocks/direction -------
// block = (dir, bt): 8 waves, wave = ht. W_h lives in VGPRs as B-fragments; h ping-pongs
// through 2x4KB LDS; next timestep's Gx tiles TDM-prefetched into 2x32KB LDS ping-pong.
__global__ void __launch_bounds__(256) lstm_kernel(
    const float* __restrict__ Gx_fw, const float* __restrict__ Gx_bw,
    const __bf16* __restrict__ Wth_fw, const __bf16* __restrict__ Wth_bw,
    float* __restrict__ h_out) {
  __shared__ __align__(128) float  gxbuf[2][NTN * 256];   // 2 x 32KB
  __shared__ __align__(16)  __bf16 hbuf[2][16 * HID];     // 2 x 4KB
  int blk = blockIdx.x;
  int dir = blk >> 4;
  int bt  = blk & 15;
  const float*  Gx  = dir ? Gx_bw  : Gx_fw;
  const __bf16* Wth = dir ? Wth_bw : Wth_fw;
  int tid  = threadIdx.x;
  int lane = tid & 31;
  int ht   = tid >> 5;       // wave id == hidden tile
  int hi   = (lane >= 16);

  {  // zero both h buffers (2*16*128 halves = 2048 uints)
    uint32_t* p = (uint32_t*)&hbuf[0][0];
    for (int i = tid; i < 2048; i += 256) p[i] = 0u;
  }

  // prefetch timestep 0
  if (ht == 0) {
    int t0 = dir ? (TT - 1) : 0;
    tdm_issue(Gx, t0, bt, (uint32_t)(uintptr_t)&gxbuf[0][0]);
  }

  // preload recurrent-weight B fragments: 4 gates x 4 k-chunks (128 VGPRs)
  v16bf Wf[4][4];
#pragma unroll
  for (int g = 0; g < 4; ++g)
#pragma unroll
    for (int kc = 0; kc < 4; ++kc) {
      int n  = g * HID + ht * 16 + (lane & 15);
      int ks = kc * 32 + (hi ? 16 : 0);
      Wf[g][kc] = *(const v16bf*)(Wth + (size_t)n * HID + ks);
    }

  v8f cst = {};
  for (int s = 0; s < TT; ++s) {
    if (ht == 0) __builtin_amdgcn_s_wait_tensorcnt(0);
    __syncthreads();                       // Gx[s] in LDS + h[s] visible to all waves
    if (ht == 0 && s + 1 < TT) {           // overlap next step's Gx fetch with compute
      int tn = dir ? (TT - 2 - s) : (s + 1);
      tdm_issue(Gx, tn, bt, (uint32_t)(uintptr_t)&gxbuf[(s + 1) & 1][0]);
    }
    const float*  gxc  = gxbuf[s & 1];
    const __bf16* rbuf = hbuf[s & 1];
    __bf16*       wbuf = hbuf[(s + 1) & 1];

    v8f acc[4];
#pragma unroll
    for (int g = 0; g < 4; ++g)
      acc[g] = *(const v8f*)(gxc + (g * 8 + ht) * 256 + lane * 8);
#pragma unroll
    for (int kc = 0; kc < 4; ++kc) {
      int row = lane & 15;
      int ks0 = kc * 32 + (hi ? 16 : 0);
      v8bf a0 = *(const v8bf*)(rbuf + row * HID + ks0);
      v8bf a1 = *(const v8bf*)(rbuf + row * HID + ks0 + 8);
      v16bf a = compose16(a0, a1);
#pragma unroll
      for (int g = 0; g < 4; ++g)
        acc[g] = __builtin_amdgcn_wmma_f32_16x16x32_bf16(
            false, a, false, Wf[g][kc], (short)0, acc[g], false, false);
    }
    int col = ht * 16 + (lane & 15);
#pragma unroll
    for (int r = 0; r < 8; ++r) {
      float gi = sigf(acc[0][r]);
      float gj = tanh_fast(acc[1][r]);
      float gf = sigf(acc[2][r] + 1.0f);
      float go = sigf(acc[3][r]);
      float cn = gf * cst[r] + gi * gj;
      cst[r] = cn;
      float hv = go * tanh_fast(cn);
      int rrow = r + (hi ? 8 : 0);                    // C layout: vgpr r -> row r / r+8
      wbuf[rrow * HID + col] = (__bf16)hv;
      if (s == TT - 1)
        h_out[(size_t)(bt * 16 + rrow) * 256 + dir * HID + col] = hv;
    }
  }
}

// ------- head: dense(64, elu) -> dense(6, sigmoid) -------
__global__ void head_kernel(const float* __restrict__ h_out, const float* __restrict__ W1,
                            const float* __restrict__ b1, const float* __restrict__ W2,
                            const float* __restrict__ b2, float* __restrict__ out) {
  int b = blockIdx.x * blockDim.x + threadIdx.x;
  if (b >= BB) return;
  const float* hr = h_out + (size_t)b * 256;
  float d1[64];
  for (int j = 0; j < 64; ++j) {
    float s = b1[j];
    for (int k = 0; k < 256; ++k) s += hr[k] * W1[k * 64 + j];
    d1[j] = (s > 0.0f) ? s : (__expf(s) - 1.0f);
  }
  for (int c = 0; c < 6; ++c) {
    float s = b2[c];
    for (int j = 0; j < 64; ++j) s += d1[j] * W2[j * 6 + c];
    out[b * 6 + c] = sigf(s);
  }
}

extern "C" void kernel_launch(void* const* d_in, const int* in_sizes, int n_in,
                              void* d_out, int out_size, void* d_ws, size_t ws_size,
                              hipStream_t stream) {
  (void)in_sizes; (void)n_in; (void)out_size; (void)ws_size;
  const int*   words = (const int*)d_in[0];
  const int*   caps  = (const int*)d_in[1];
  const float* wemb  = (const float*)d_in[2];
  const float* cemb  = (const float*)d_in[3];
  const float* W_fw  = (const float*)d_in[4];
  const float* b_fw  = (const float*)d_in[5];
  const float* W_bw  = (const float*)d_in[6];
  const float* b_bw  = (const float*)d_in[7];
  const float* W1    = (const float*)d_in[8];
  const float* b1    = (const float*)d_in[9];
  const float* W2    = (const float*)d_in[10];
  const float* b2    = (const float*)d_in[11];

  char* ws = (char*)d_ws;
  size_t off = 0;
  auto alloc = [&](size_t bytes) -> void* {
    void* p = ws + off;
    off += (bytes + 255) & ~(size_t)255;
    return p;
  };
  __bf16* X      = (__bf16*)alloc((size_t)TT * BB * KX * 2);
  __bf16* Wtx_fw = (__bf16*)alloc((size_t)G4 * KX * 2);
  __bf16* Wtx_bw = (__bf16*)alloc((size_t)G4 * KX * 2);
  __bf16* Wth_fw = (__bf16*)alloc((size_t)G4 * HID * 2);
  __bf16* Wth_bw = (__bf16*)alloc((size_t)G4 * HID * 2);
  float*  Gx_fw  = (float*)alloc((size_t)TT * NTN * NTM * 256 * 4);
  float*  Gx_bw  = (float*)alloc((size_t)TT * NTN * NTM * 256 * 4);
  float*  h_out  = (float*)alloc((size_t)BB * 256 * 4);

  embed_kernel<<<TT * BB, KX, 0, stream>>>(words, caps, wemb, cemb, X);

  int pwThreads = G4 * (KX + HID);
  prepw_kernel<<<(pwThreads + 255) / 256, 256, 0, stream>>>(W_fw, Wtx_fw, Wth_fw);
  prepw_kernel<<<(pwThreads + 255) / 256, 256, 0, stream>>>(W_bw, Wtx_bw, Wth_bw);

  gemmx_kernel<<<4000, 256, 0, stream>>>(X, Wtx_fw, b_fw, Gx_fw);
  gemmx_kernel<<<4000, 256, 0, stream>>>(X, Wtx_bw, b_bw, Gx_bw);

  // 16 independent batch-tiles per direction
  lstm_kernel<<<32, 256, 0, stream>>>(Gx_fw, Gx_bw, Wth_fw, Wth_bw, h_out);

  head_kernel<<<1, 256, 0, stream>>>(h_out, W1, b1, W2, b2, (float*)d_out);
}
